// GML_39170101739975
// MI455X (gfx1250) — matrix operand order repaired and checked by reading, not compiled, not run on hardware
//
#include <hip/hip_runtime.h>
#include <hip/hip_bf16.h>

typedef __attribute__((ext_vector_type(16))) _Float16 v16h;
typedef __attribute__((ext_vector_type(8)))  _Float16 v8h;
typedef __attribute__((ext_vector_type(8)))  float    v8f;
typedef __attribute__((ext_vector_type(4)))  float    v4f;

#define D_EMB 512

__global__ void GML_zero_out(float* out) { out[0] = 0.0f; }

// Phase 1: Y[n][m*D + d] = (f16) normalize(multihead[m] * embedding[n])
// One wave32 per (node, head); lane handles 16 contiguous dims.
__global__ void GML_normalize(const float* __restrict__ emb,
                              const float* __restrict__ mh,
                              _Float16* __restrict__ Y,
                              int N, int M) {
  int wave = (int)((blockIdx.x * blockDim.x + threadIdx.x) >> 5);
  int lane = (int)(threadIdx.x & 31);
  if (wave >= N * M) return;
  int n = wave / M, m = wave % M;

  const float* e = emb + (size_t)n * D_EMB + lane * 16;
  const float* w = mh  + (size_t)m * D_EMB + lane * 16;

  float x[16];
  float ss = 0.0f;
  #pragma unroll
  for (int i = 0; i < 4; ++i) {
    v4f ev = *(const v4f*)(e + i * 4);
    v4f wv = *(const v4f*)(w + i * 4);
    #pragma unroll
    for (int j = 0; j < 4; ++j) {
      float p = ev[j] * wv[j];
      x[i * 4 + j] = p;
      ss = fmaf(p, p, ss);
    }
  }
  #pragma unroll
  for (int off = 16; off >= 1; off >>= 1)
    ss += __shfl_xor(ss, off, 32);

  float inv = 1.0f / fmaxf(sqrtf(ss), 1e-12f);   // F.normalize eps

  _Float16* yo = Y + (size_t)n * ((size_t)M * D_EMB) + (size_t)m * D_EMB + lane * 16;
  v8h o0, o1;
  #pragma unroll
  for (int j = 0; j < 8; ++j) {
    o0[j] = (_Float16)(x[j] * inv);
    o1[j] = (_Float16)(x[8 + j] * inv);
  }
  *(v8h*)yo       = o0;
  *(v8h*)(yo + 8) = o1;
}

// Phase 2: one wave32 per 16 edges. WMMA accumulates a 16x16 Gram tile of
// (src rows) x (dst rows); diagonal = per-edge dot over K = M*D dims.
__global__ void GML_edge_loss(const int* __restrict__ pos,
                              const int* __restrict__ neg,
                              const _Float16* __restrict__ Y,
                              float* __restrict__ out,
                              int E, int K, float invM) {
  int wave = (int)((blockIdx.x * blockDim.x + threadIdx.x) >> 5);
  int lane = (int)(threadIdx.x & 31);
  int gp = (E + 15) >> 4;          // positive-edge groups
  if (wave >= 2 * gp) return;
  bool is_neg = (wave >= gp);
  const int* ep = is_neg ? neg : pos;
  int base = (is_neg ? (wave - gp) : wave) << 4;

  int j = lane & 15;               // edge slot within tile (also A row / B col)
  int e = base + j;
  int ec = (e < E) ? e : (E - 1);
  int si = ep[ec];
  int di = ep[(size_t)E + ec];
  const _Float16* rowA = Y + (size_t)si * K;   // A row j (source node)
  const _Float16* rowB = Y + (size_t)di * K;   // B col j (dest node)
  int grp = lane >> 4;

  v8f c = {};
  for (int kb = 0; kb < K; kb += 32) {
    // A 16x32 layout: halves 0-7 -> K = kb+grp*8.., halves 8-15 -> K = kb+16+grp*8..
    v8h alo = *(const v8h*)(rowA + kb + grp * 8);
    v8h ahi = *(const v8h*)(rowA + kb + 16 + grp * 8);
    // B 32x16 layout: halves 0-7 -> K = kb+grp*8.., halves 8-15 -> K = kb+16+grp*8..
    v8h blo = *(const v8h*)(rowB + kb + grp * 8);
    v8h bhi = *(const v8h*)(rowB + kb + 16 + grp * 8);
    v16h a = __builtin_shufflevector(alo, ahi, 0,1,2,3,4,5,6,7,8,9,10,11,12,13,14,15);
    v16h b = __builtin_shufflevector(blo, bhi, 0,1,2,3,4,5,6,7,8,9,10,11,12,13,14,15);
    c = __builtin_amdgcn_wmma_f32_16x16x32_f16(
        /*neg_a=*/false, a, /*neg_b=*/false, b,
        /*c_mod=*/(short)0, c, /*reuse_a=*/false, /*reuse_b=*/false);
  }

  // Diagonal D[i][i]: lanes 0-7 (VGPR=lane) and lanes 24-31 (VGPR=lane-24 ... = lane%8)
  bool active = (lane < 8) || (lane >= 24);
  int i = (lane < 8) ? lane : (lane - 16);     // diag index 0..15
  float term = 0.0f;
  if (active && (base + i) < E) {
    int r = lane & 7;
    float dot = 0.0f;
    #pragma unroll
    for (int q = 0; q < 8; ++q) if (q == r) dot = c[q];
    float dist = (1.0f + 1e-6f) - dot * invM;
    term = is_neg ? logf(1.0f - 0.5f * dist) : logf(dist);
  }
  #pragma unroll
  for (int off = 16; off >= 1; off >>= 1)
    term += __shfl_xor(term, off, 32);
  if (lane == 0) atomicAdd(out, -term);        // loss = -(sum of log terms)
}

extern "C" void kernel_launch(void* const* d_in, const int* in_sizes, int n_in,
                              void* d_out, int out_size, void* d_ws, size_t ws_size,
                              hipStream_t stream) {
  const int*   edges     = (const int*)d_in[0];
  const int*   neg_edges = (const int*)d_in[1];
  const float* embedding = (const float*)d_in[2];
  const float* multihead = (const float*)d_in[3];
  float* out = (float*)d_out;

  const int E = in_sizes[0] / 2;
  const int N = in_sizes[2] / D_EMB;
  const int M = in_sizes[3] / D_EMB;
  const int K = M * D_EMB;

  _Float16* Y = (_Float16*)d_ws;   // N * K f16 (= 32 MB at N=8192, K=2048) fits in L2

  GML_zero_out<<<1, 1, 0, stream>>>(out);

  {
    int waves  = N * M;
    int blocks = (waves + 7) / 8;              // 256 thr = 8 wave32/WG
    GML_normalize<<<blocks, 256, 0, stream>>>(embedding, multihead, Y, N, M);
  }
  {
    int gp     = (E + 15) >> 4;
    int waves  = 2 * gp;
    int blocks = (waves + 7) / 8;
    GML_edge_loss<<<blocks, 256, 0, stream>>>(edges, neg_edges, Y, out,
                                              E, K, 1.0f / (float)M);
  }
}
